// GraphPromptTuningLoss_13365938225381
// MI455X (gfx1250) — compile-verified, matrix operand order checked by reading, not run on hardware
//
#include <hip/hip_runtime.h>
#include <hip/hip_bf16.h>

// ---------------------------------------------------------------------------
// Cosine-similarity cross-entropy loss:
//   loss = mean_n [ -log_softmax( (e_n/||e_n||) . (c_k/||c_k||) / tau )[label_n] ]
// N=65536, K=512, D=1024, tau=0.1
//
// MI455X plan:
//  * v_wmma_f32_16x16x32_bf16 GEMM, M-tile = 32 rows/block (halves L2 B-traffic
//    vs M=16: 2048 blocks x 1MB = 2GB; each B fragment feeds 2 WMMAs).
//  * A (embeddings) streamed once from HBM (256MB ~ 11us floor), converted
//    fp32->bf16 in-register with v_perm_b32 pair-packing (1 perm / 2 elems).
//  * Row ||e||^2 accumulated in f32 during conversion; scale applied post-GEMM.
//  * Centers pre-normalized to bf16 in d_ws (1MB, L2-resident).
//  * log-softmax + NLL epilogue in LDS, two 16-row passes (LDS ~33KB),
//    one f32 atomic per block.
// ---------------------------------------------------------------------------

typedef __attribute__((ext_vector_type(16))) __bf16 v16bf;
typedef __attribute__((ext_vector_type(8)))  float  v8f;

#define N_ROWS    65536
#define K_CENTERS 512
#define D_DIM     1024
#define M_BLOCK   32
#define TAU_INV   10.0f
#define CEPS      1e-8f
#define LDS_STRIDE 516   // 512 + 4 pad: rows 8 apart land on different banks

static __device__ __forceinline__ unsigned short f32_to_bf16_rne(float f) {
    unsigned int u = __float_as_uint(f);
    unsigned int r = u + 0x7FFFu + ((u >> 16) & 1u);
    return (unsigned short)(r >> 16);
}

// pack two fp32 -> two bf16 (round-to-nearest) in one u32: low=x, high=y.
// v_perm_b32 selects the two rounded high halfwords in a single VALU op.
static __device__ __forceinline__ unsigned int pack_bf16_rn(float x, float y) {
    unsigned int ux = __float_as_uint(x) + 0x8000u;
    unsigned int uy = __float_as_uint(y) + 0x8000u;
    // result bytes: [0]=ux.b2 [1]=ux.b3 [2]=uy.b2 [3]=uy.b3
    return __builtin_amdgcn_perm(uy, ux, 0x07060302u);
}

// ------------------------- center normalization ----------------------------
__global__ __launch_bounds__(256)
void normalize_centers_kernel(const float* __restrict__ centers,
                              unsigned short* __restrict__ cb) {
    __shared__ float red[8];
    const int k = blockIdx.x;        // center index 0..511
    const int t = threadIdx.x;       // 0..255, 4 floats each
    const float* row = centers + (size_t)k * D_DIM;

    float4 v = ((const float4*)row)[t];
    float ss = v.x * v.x + v.y * v.y + v.z * v.z + v.w * v.w;
    #pragma unroll
    for (int off = 16; off > 0; off >>= 1) ss += __shfl_xor(ss, off, 32);
    if ((t & 31) == 0) red[t >> 5] = ss;
    __syncthreads();
    float tot = 0.f;
    #pragma unroll
    for (int i = 0; i < 8; ++i) tot += red[i];

    const float inv = 1.0f / fmaxf(sqrtf(tot), CEPS);

    // keep full RNE here (runs once over 2MB, cost irrelevant)
    unsigned int p0 = (unsigned int)f32_to_bf16_rne(v.x * inv) |
                      ((unsigned int)f32_to_bf16_rne(v.y * inv) << 16);
    unsigned int p1 = (unsigned int)f32_to_bf16_rne(v.z * inv) |
                      ((unsigned int)f32_to_bf16_rne(v.w * inv) << 16);
    ((uint2*)(cb + (size_t)k * D_DIM))[t] = make_uint2(p0, p1);
}

// load 16 fp32 of one A-row chunk, accumulate ||.||^2, emit bf16 fragment.
// CDNA5 16-bit A 16x32 layout: lane-half 0 holds K = d..d+7, d+16..d+23;
// lane-half 1 holds +8 (dbase already includes khalf*8).
static __device__ __forceinline__ v16bf load_convert_a(const float* __restrict__ p,
                                                       float& sumsq) {
    float4 a0 = *(const float4*)(p);
    float4 a1 = *(const float4*)(p + 4);
    float4 a2 = *(const float4*)(p + 16);
    float4 a3 = *(const float4*)(p + 20);
    sumsq += a0.x*a0.x + a0.y*a0.y + a0.z*a0.z + a0.w*a0.w
           + a1.x*a1.x + a1.y*a1.y + a1.z*a1.z + a1.w*a1.w
           + a2.x*a2.x + a2.y*a2.y + a2.z*a2.z + a2.w*a2.w
           + a3.x*a3.x + a3.y*a3.y + a3.z*a3.z + a3.w*a3.w;
    union { v16bf v; unsigned int ui[8]; } au;
    au.ui[0] = pack_bf16_rn(a0.x, a0.y);
    au.ui[1] = pack_bf16_rn(a0.z, a0.w);
    au.ui[2] = pack_bf16_rn(a1.x, a1.y);
    au.ui[3] = pack_bf16_rn(a1.z, a1.w);
    au.ui[4] = pack_bf16_rn(a2.x, a2.y);
    au.ui[5] = pack_bf16_rn(a2.z, a2.w);
    au.ui[6] = pack_bf16_rn(a3.x, a3.y);
    au.ui[7] = pack_bf16_rn(a3.z, a3.w);
    return au.v;
}

// ---------------------- fused GEMM + log-softmax + NLL ---------------------
__global__ __launch_bounds__(256)
void cosine_ce_kernel(const float* __restrict__ emb,
                      const unsigned short* __restrict__ cb,
                      const long long* __restrict__ labels,
                      float* __restrict__ out) {
    __shared__ float tileL[16 * LDS_STRIDE];  // one 16-row half at a time
    __shared__ float scaleL[M_BLOCK];         // TAU_INV / max(||e||, eps)
    __shared__ float lossL[M_BLOCK];

    const int tid   = threadIdx.x;
    const int lane  = tid & 31;
    const int wave  = tid >> 5;          // 0..7, each owns 64 output columns
    const int lrow  = lane & 15;         // A-row / B-col inside a 16x16 tile
    const int khalf = lane >> 4;         // which K-half this lane holds
    const int rowbase = blockIdx.x * M_BLOCK;

    const float* __restrict__ arow0 = emb + (size_t)(rowbase + lrow) * D_DIM;
    const float* __restrict__ arow1 = arow0 + (size_t)16 * D_DIM;

    v8f acc[2][4] = {};                  // [M-tile][N-tile], 16x16 f32 each
    float sumsq0 = 0.f, sumsq1 = 0.f;

    #pragma unroll 1
    for (int dk = 0; dk < D_DIM; dk += 32) {
        const int dbase = dk + khalf * 8;

        __builtin_prefetch(arow0 + dk + 64, 0, 1);   // streamed-once A rows
        __builtin_prefetch(arow1 + dk + 64, 0, 1);

        const v16bf a0 = load_convert_a(arow0 + dbase, sumsq0);
        const v16bf a1 = load_convert_a(arow1 + dbase, sumsq1);

        #pragma unroll
        for (int t = 0; t < 4; ++t) {
            const int col = wave * 64 + t * 16 + lrow;
            const unsigned short* bp = cb + (size_t)col * D_DIM + dbase;
            union { v16bf v; uint4 q[2]; } bu;
            bu.q[0] = *(const uint4*)bp;          // K = d..d+7   (8 bf16)
            bu.q[1] = *(const uint4*)(bp + 16);   // K = d+16..d+23
            acc[0][t] = __builtin_amdgcn_wmma_f32_16x16x32_bf16(
                            false, a0, false, bu.v, (short)0, acc[0][t], false, false);
            acc[1][t] = __builtin_amdgcn_wmma_f32_16x16x32_bf16(
                            false, a1, false, bu.v, (short)0, acc[1][t], false, false);
        }
    }

    // full row sum-of-squares: combine the two K-halves (lane <-> lane^16)
    sumsq0 += __shfl_xor(sumsq0, 16, 32);
    sumsq1 += __shfl_xor(sumsq1, 16, 32);
    if (wave == 0 && khalf == 0) {
        scaleL[lrow]      = TAU_INV / fmaxf(sqrtf(sumsq0), CEPS);
        scaleL[lrow + 16] = TAU_INV / fmaxf(sqrtf(sumsq1), CEPS);
    }

    // two 16-row epilogue passes over one 33KB LDS tile
    #pragma unroll 1
    for (int half = 0; half < 2; ++half) {
        // spill: C layout -> lanes 0-15: N=lane, M=j ; lanes 16-31: M=j+8
        #pragma unroll
        for (int t = 0; t < 4; ++t) {
            const int col = wave * 64 + t * 16 + lrow;
            #pragma unroll
            for (int j = 0; j < 8; ++j) {
                const int row = j + khalf * 8;
                tileL[row * LDS_STRIDE + col] = acc[half][t][j];
            }
        }
        __syncthreads();

        // log-softmax + NLL: 2 rows per wave, 16 cols per lane
        #pragma unroll
        for (int rr = 0; rr < 2; ++rr) {
            const int r  = wave * 2 + rr;          // local row in this half
            const int gr = half * 16 + r;          // row in block
            const float sc = scaleL[gr];
            float vals[16];
            float m = -3.402823466e+38f;
            #pragma unroll
            for (int i = 0; i < 16; ++i) {
                float v = tileL[r * LDS_STRIDE + lane + 32 * i] * sc;
                vals[i] = v;
                m = fmaxf(m, v);
            }
            #pragma unroll
            for (int off = 16; off > 0; off >>= 1) m = fmaxf(m, __shfl_xor(m, off, 32));
            float s = 0.f;
            #pragma unroll
            for (int i = 0; i < 16; ++i) s += __expf(vals[i] - m);
            #pragma unroll
            for (int off = 16; off > 0; off >>= 1) s += __shfl_xor(s, off, 32);
            if (lane == 0) {
                const long long lab = labels[rowbase + gr];
                const float ll = tileL[r * LDS_STRIDE + (int)lab] * sc;
                lossL[gr] = __logf(s) + m - ll;
            }
        }
        __syncthreads();   // protect tileL before next half overwrites it
    }

    if (tid == 0) {
        float blk = 0.f;
        #pragma unroll
        for (int i = 0; i < M_BLOCK; ++i) blk += lossL[i];
        atomicAdd(out, blk * (1.0f / (float)N_ROWS));
    }
}

// ---------------------------------------------------------------------------
extern "C" void kernel_launch(void* const* d_in, const int* in_sizes, int n_in,
                              void* d_out, int out_size, void* d_ws, size_t ws_size,
                              hipStream_t stream) {
    const float*      emb = (const float*)d_in[0];       // [65536,1024] f32
    const float*      ctr = (const float*)d_in[1];       // [512,1024] f32
    const long long*  lab = (const long long*)d_in[2];   // [65536] i64
    float*            out = (float*)d_out;               // scalar f32
    unsigned short*   cb  = (unsigned short*)d_ws;       // [512,1024] bf16 (1MB)

    hipMemsetAsync(d_out, 0, sizeof(float), stream);
    normalize_centers_kernel<<<K_CENTERS, 256, 0, stream>>>(ctr, cb);
    cosine_ce_kernel<<<N_ROWS / M_BLOCK, 256, 0, stream>>>(emb, cb, lab, out);
}
